// MultiHeadAttention_9328668967348
// MI455X (gfx1250) — matrix operand order repaired and checked by reading, not compiled, not run on hardware
//
#include <hip/hip_runtime.h>
#include <hip/hip_bf16.h>

// ---------------- problem constants (match reference) ----------------
#define BATCH   2
#define SEQ     4096
#define NHEAD   8
#define DK      64
#define DMODEL  512
#define MROWS   (BATCH * SEQ)      // 8192
#define ROWP    (SEQ + 4)          // padded LDS row stride (16B-aligned rows)

// ---------------- CDNA5 WMMA types ----------------
typedef __attribute__((ext_vector_type(16))) __bf16 v16bf;
typedef __attribute__((ext_vector_type(8)))  float  v8f;

static __device__ inline v8f zero8() {
    v8f z = {0.f,0.f,0.f,0.f,0.f,0.f,0.f,0.f}; return z;
}
static __device__ inline v8f wmma_bf16(v16bf a, v16bf b, v8f c) {
    return __builtin_amdgcn_wmma_f32_16x16x32_bf16(
        false, a, false, b, (short)0, c, false, false);
}

// Pack a 16-element bf16 fragment from two contiguous 8-float runs
// (CDNA5 16-bit A/B per-lane K-striping: VGPRs 0-3 = src[khalf..+7],
//  VGPRs 4-7 = src[16+khalf..+7]).  32B runs -> float4 vector loads + cvt.
static __device__ inline v16bf pack2x8(const float* __restrict__ p0,
                                       const float* __restrict__ p1,
                                       float s) {
    const float4* q0 = (const float4*)p0;
    const float4* q1 = (const float4*)p1;
    float4 x0 = q0[0], x1 = q0[1], y0 = q1[0], y1 = q1[1];
    v16bf r;
    r[0]=(__bf16)(x0.x*s); r[1]=(__bf16)(x0.y*s); r[2]=(__bf16)(x0.z*s); r[3]=(__bf16)(x0.w*s);
    r[4]=(__bf16)(x1.x*s); r[5]=(__bf16)(x1.y*s); r[6]=(__bf16)(x1.z*s); r[7]=(__bf16)(x1.w*s);
    r[8]=(__bf16)(y0.x*s); r[9]=(__bf16)(y0.y*s); r[10]=(__bf16)(y0.z*s); r[11]=(__bf16)(y0.w*s);
    r[12]=(__bf16)(y1.x*s); r[13]=(__bf16)(y1.y*s); r[14]=(__bf16)(y1.z*s); r[15]=(__bf16)(y1.w*s);
    return r;
}

// 16 pre-converted bf16 as two contiguous 8-element (16B) runs -> two b128
// loads, zero VALU.  Works for global or LDS pointers.
static __device__ inline v16bf load2x8bf(const unsigned short* __restrict__ p0,
                                         const unsigned short* __restrict__ p1) {
    union { v16bf v; uint4 q[2]; } u;
    u.q[0] = *(const uint4*)p0;
    u.q[1] = *(const uint4*)p1;
    return u.v;
}

static __device__ inline unsigned short bfbits(float f) {
    __bf16 h = (__bf16)f;
    return __builtin_bit_cast(unsigned short, h);
}

// =====================================================================
// GEMM:  Y = X (*xscale) @ W  (+ residual), M=8192, N=K=512.
// block = 256 threads (8 waves); tile = 128 rows x 64 cols.
// W chunk staged in LDS transposed [n=64][k=32] bf16 -> contiguous B frags.
// Compile-time variants:
//   XBF16   : X is bf16 (raw b128 fragment loads, no cvt)
//   OUT_MODE: 0 = f32 row-major [M,512]
//             1 = bf16 head-major [B,H,S,64]
//             2 = bf16 d-major   [B,H,64,S]  (for V -> contiguous attn@V B)
//   HAS_RES : add f32 residual (row-major [M,512])
// =====================================================================
template <int XBF16, int OUT_MODE, int HAS_RES>
__global__ __launch_bounds__(256)
void gemm_bf16(const void* __restrict__ Xv,
               const float* __restrict__ W,
               const float* __restrict__ residual,
               void* __restrict__ Yv,
               float xscale)
{
    __shared__ unsigned short wlds[64 * 32];         // 4 KB bf16, [n][k]

    const int wave = threadIdx.x >> 5;
    const int lane = threadIdx.x & 31;
    const int half = (lane >> 4) & 1;
    const int l16  = lane & 15;
    const int r0   = blockIdx.x * 128 + wave * 16;
    const int c0   = blockIdx.y * 64;
    const int khalf = half * 8;
    const int kb    = half * 16;

    v8f acc[4];
#pragma unroll
    for (int i = 0; i < 4; ++i) acc[i] = zero8();

    for (int kk = 0; kk < DMODEL; kk += 32) {
        {   // stage W[kk:kk+32, c0:c0+64] -> LDS bf16, transposed
            int base = threadIdx.x * 8;              // coalesced global reads
#pragma unroll
            for (int i = 0; i < 8; ++i) {
                int e  = base + i;
                int kr = e >> 6, nc = e & 63;
                wlds[nc * 32 + kr] =
                    bfbits(W[(size_t)(kk + kr) * DMODEL + c0 + nc]);
            }
        }
        __syncthreads();

        v16bf a;
        if constexpr (XBF16) {
            const unsigned short* ap =
                (const unsigned short*)Xv + (size_t)(r0 + l16) * DMODEL + kk;
            a = load2x8bf(ap + khalf, ap + 16 + khalf);
        } else {
            const float* ap = (const float*)Xv + (size_t)(r0 + l16) * DMODEL + kk;
            a = pack2x8(ap + khalf, ap + 16 + khalf, xscale);
        }

#pragma unroll
        for (int nt = 0; nt < 4; ++nt) {
            v16bf b = load2x8bf(&wlds[(nt * 16 + l16) * 32 + khalf],
                                &wlds[(nt * 16 + l16) * 32 + 16 + khalf]);
            acc[nt] = wmma_bf16(a, b, acc[nt]);
        }
        __syncthreads();
    }

    if constexpr (OUT_MODE == 2) {
        // stage the 128x64 tile in LDS, write out d-major bf16, coalesced
        __shared__ float tbuf[128][65];              // +1 pad: conflict-free
#pragma unroll
        for (int nt = 0; nt < 4; ++nt)
#pragma unroll
            for (int r = 0; r < 8; ++r)
                tbuf[wave * 16 + r + half * 8][nt * 16 + l16] = acc[nt][r];
        __syncthreads();

        int b_ = (blockIdx.x * 128) >> 12;
        int s0 = (blockIdx.x * 128) & (SEQ - 1);
        int h  = blockIdx.y;
        unsigned short* yt =
            (unsigned short*)Yv + (((size_t)b_ * NHEAD + h) * DK) * SEQ;
        for (int e = threadIdx.x; e < 128 * 64; e += 256) {
            int d = e >> 7, sl = e & 127;            // coalesced along s
            yt[(size_t)d * SEQ + s0 + sl] = bfbits(tbuf[sl][d]);
        }
    } else {
#pragma unroll
        for (int nt = 0; nt < 4; ++nt) {
#pragma unroll
            for (int r = 0; r < 8; ++r) {
                int row = r0 + r + half * 8;
                int col = c0 + nt * 16 + l16;
                float v = acc[nt][r];
                if constexpr (HAS_RES)
                    v += residual[(size_t)row * DMODEL + col];
                if constexpr (OUT_MODE == 1) {
                    int bi = row >> 12, s = row & (SEQ - 1);
                    int h  = col >> 6,  d = col & 63;
                    ((unsigned short*)Yv)
                        [((((size_t)bi * NHEAD + h) * SEQ) + s) * DK + d] = bfbits(v);
                } else {
                    ((float*)Yv)[(size_t)row * DMODEL + col] = v;
                }
            }
        }
    }
}

// =====================================================================
// Fused attention: one workgroup = (b, h, 16 query rows).
// 16 x 4096 fp32 score slab lives in LDS (≈256 KB of the 320 KB WGP LDS);
// the 1.07 GB attn tensor is written to HBM exactly once, never re-read.
// Q/K/V arrive pre-converted bf16 -> all WMMA fragments are raw b128 loads.
// =====================================================================
__global__ __launch_bounds__(256)
void attn_fused(const unsigned short* __restrict__ qh, // [B*H,S,64] bf16, pre-scaled
                const unsigned short* __restrict__ kh, // [B*H,S,64] bf16
                const unsigned short* __restrict__ vt, // [B*H,64,S] bf16 (transposed)
                float* __restrict__ attn_g,            // [B*H,S,S] f32
                unsigned short* __restrict__ outh)     // [B*S,512] bf16
{
    extern __shared__ float sm[];               // 16 * ROWP floats

    const int wave = threadIdx.x >> 5;
    const int lane = threadIdx.x & 31;
    const int half = (lane >> 4) & 1;
    const int l16  = lane & 15;
    const int khalf = half * 8;

    const int bh = blockIdx.x >> 8;             // 256 q-tiles per (b,h)
    const int q0 = (blockIdx.x & 255) * 16;

    const unsigned short* Q  = qh + (size_t)bh * SEQ * DK;
    const unsigned short* K  = kh + (size_t)bh * SEQ * DK;
    const unsigned short* Vt = vt + (size_t)bh * DK * SEQ;

    // ---- preload Q tile: two 16x32 bf16 A fragments (dk = 64) ----
    const unsigned short* qp = Q + (size_t)(q0 + l16) * DK;
    v16bf aq0 = load2x8bf(qp +      half * 16, qp +  8 + half * 16);
    v16bf aq1 = load2x8bf(qp + 32 + half * 16, qp + 40 + half * 16);

    // ---- phase 1: logits into LDS ----
    for (int t = 0; t < 32; ++t) {
        int s0 = t * 128 + wave * 16;           // this wave's 16-key tile
        const unsigned short* kp = K + (size_t)(s0 + l16) * DK + half * 16;
        v16bf b0 = load2x8bf(kp,      kp + 8);
        v16bf b1 = load2x8bf(kp + 32, kp + 40);
        v8f c = zero8();
        c = wmma_bf16(aq0, b0, c);
        c = wmma_bf16(aq1, b1, c);
#pragma unroll
        for (int r = 0; r < 8; ++r)
            sm[(r + half * 8) * ROWP + s0 + l16] = c[r];
    }
    __syncthreads();

    // ---- phase 2: softmax, wave w owns rows 2w, 2w+1 (float4 passes) ----
#pragma unroll
    for (int rr = 0; rr < 2; ++rr) {
        int row = wave * 2 + rr;
        float4* rp4 = (float4*)(sm + row * ROWP);
        float m = -3.4e38f;
        for (int i = lane; i < SEQ / 4; i += 32) {
            float4 x = rp4[i];
            m = fmaxf(m, fmaxf(fmaxf(x.x, x.y), fmaxf(x.z, x.w)));
        }
#pragma unroll
        for (int o = 16; o > 0; o >>= 1) m = fmaxf(m, __shfl_xor(m, o, 32));
        float s = 0.f;
        for (int i = lane; i < SEQ / 4; i += 32) {
            float4 x = rp4[i];
            x.x = __expf(x.x - m); x.y = __expf(x.y - m);
            x.z = __expf(x.z - m); x.w = __expf(x.w - m);
            rp4[i] = x;
            s += (x.x + x.y) + (x.z + x.w);
        }
#pragma unroll
        for (int o = 16; o > 0; o >>= 1) s += __shfl_xor(s, o, 32);
        float inv = 1.f / s;
        float4* gp4 = (float4*)(attn_g + ((size_t)bh * SEQ + q0 + row) * SEQ);
        for (int i = lane; i < SEQ / 4; i += 32) {
            float4 x = rp4[i];
            x.x *= inv; x.y *= inv; x.z *= inv; x.w *= inv;
            rp4[i] = x;
            gp4[i] = x;                         // coalesced b128, written once
        }
    }
    __syncthreads();

    // ---- phase 3: out = attn @ V, wave w owns keys [512w, 512w+512) ----
    v8f acc[4];
#pragma unroll
    for (int i = 0; i < 4; ++i) acc[i] = zero8();

    for (int cix = 0; cix < 16; ++cix) {
        int kb = wave * 512 + cix * 32;
        const float* ap = sm + l16 * ROWP + kb;  // attn row (ds_load_b128 x4)
        v16bf a = pack2x8(ap + half * 16, ap + half * 16 + 8, 1.f);
#pragma unroll
        for (int nt = 0; nt < 4; ++nt) {
            int d = nt * 16 + l16;
            // Vt[d][kb+half*16 .. +15]: 16 contiguous bf16 -> b128 x2
            const unsigned short* vp = Vt + (size_t)d * SEQ + kb + half * 16;
            v16bf b = load2x8bf(vp, vp + 8);
            acc[nt] = wmma_bf16(a, b, acc[nt]);
        }
    }
    __syncthreads();                            // all waves done reading sm

    // cross-wave reduce through LDS (reuse score slab)
#pragma unroll
    for (int nt = 0; nt < 4; ++nt)
#pragma unroll
        for (int r = 0; r < 8; ++r)
            sm[wave * 1024 + (r + half * 8) * 64 + nt * 16 + l16] = acc[nt][r];
    __syncthreads();

    int b_ = bh >> 3, h = bh & 7;
    for (int idx = threadIdx.x; idx < 1024; idx += 256) {
        float v = 0.f;
#pragma unroll
        for (int w = 0; w < 8; ++w) v += sm[w * 1024 + idx];
        int row = idx >> 6, d = idx & 63;
        outh[((size_t)b_ * SEQ + q0 + row) * DMODEL + h * DK + d] = bfbits(v);
    }
}

// =====================================================================
// LayerNorm over last dim (512), in place, float4-vectorized. wave-per-row.
// =====================================================================
__global__ __launch_bounds__(256)
void layernorm_rows(float* __restrict__ Y,
                    const float* __restrict__ gamma,
                    const float* __restrict__ beta)
{
    int wave = threadIdx.x >> 5, lane = threadIdx.x & 31;
    int row  = blockIdx.x * 8 + wave;
    float4* rp4 = (float4*)(Y + (size_t)row * DMODEL);
    const float4* g4 = (const float4*)gamma;
    const float4* b4 = (const float4*)beta;

    float4 x[4];
    float s = 0.f, s2 = 0.f;
#pragma unroll
    for (int i = 0; i < 4; ++i) {
        float4 v = rp4[i * 32 + lane];
        x[i] = v;
        s  += (v.x + v.y) + (v.z + v.w);
        s2 += (v.x * v.x + v.y * v.y) + (v.z * v.z + v.w * v.w);
    }
#pragma unroll
    for (int o = 16; o > 0; o >>= 1) {
        s  += __shfl_xor(s,  o, 32);
        s2 += __shfl_xor(s2, o, 32);
    }
    float mean = s * (1.f / DMODEL);
    float var  = s2 * (1.f / DMODEL) - mean * mean;
    float rstd = rsqrtf(var + 1e-6f);
#pragma unroll
    for (int i = 0; i < 4; ++i) {
        float4 g = g4[i * 32 + lane], b = b4[i * 32 + lane], v = x[i];
        v.x = (v.x - mean) * rstd * g.x + b.x;
        v.y = (v.y - mean) * rstd * g.y + b.y;
        v.z = (v.z - mean) * rstd * g.z + b.z;
        v.w = (v.w - mean) * rstd * g.w + b.w;
        rp4[i * 32 + lane] = v;
    }
}

// =====================================================================
extern "C" void kernel_launch(void* const* d_in, const int* in_sizes, int n_in,
                              void* d_out, int out_size, void* d_ws, size_t ws_size,
                              hipStream_t stream) {
    const float* q     = (const float*)d_in[0];
    const float* k     = (const float*)d_in[1];
    const float* v     = (const float*)d_in[2];
    const float* Wq    = (const float*)d_in[3];
    const float* Wk    = (const float*)d_in[4];
    const float* Wv    = (const float*)d_in[5];
    const float* Wo    = (const float*)d_in[6];
    const float* gamma = (const float*)d_in[7];
    const float* beta  = (const float*)d_in[8];

    unsigned short* ws = (unsigned short*)d_ws;
    const size_t HSZ = (size_t)BATCH * NHEAD * SEQ * DK;   // 4,194,304 elems
    unsigned short* qh = ws;                               // bf16 [B,H,S,64]
    unsigned short* kh = ws + HSZ;                         // bf16 [B,H,S,64]
    unsigned short* vt = ws + 2 * HSZ;                     // bf16 [B,H,64,S]
    unsigned short* ao = ws + 3 * HSZ;                     // bf16 [B*S,512]

    float* out    = (float*)d_out;                         // [B*S, 512]
    float* attn_g = out + (size_t)BATCH * SEQ * DMODEL;    // [B,H,S,S]

    const float inv_sqrt_dk = 0.125f;                      // 1/sqrt(64)
    dim3 gblk(MROWS / 128, DMODEL / 64);                   // (64, 8)

    gemm_bf16<0,1,0><<<gblk, 256, 0, stream>>>(q, Wq, nullptr, qh, inv_sqrt_dk);
    gemm_bf16<0,1,0><<<gblk, 256, 0, stream>>>(k, Wk, nullptr, kh, 1.0f);
    gemm_bf16<0,2,0><<<gblk, 256, 0, stream>>>(v, Wv, nullptr, vt, 1.0f);

    attn_fused<<<BATCH * NHEAD * (SEQ / 16), 256,
                 (size_t)16 * ROWP * sizeof(float), stream>>>(qh, kh, vt, attn_g, ao);

    gemm_bf16<1,0,1><<<gblk, 256, 0, stream>>>(ao, Wo, q, out, 1.0f);
    layernorm_rows<<<MROWS / 8, 256, 0, stream>>>(out, gamma, beta);
}
